// S_OnsagerNet_45449343926326
// MI455X (gfx1250) — matrix operand order repaired and checked by reading, not compiled, not run on hardware
//
#include <hip/hip_runtime.h>

// ---------------------------------------------------------------------------
// OnsagerNet fused SDE step for MI455X (gfx1250, wave32, WMMA bf16 path)
//
//   B=65536 samples, N=32, H=128. All GEMMs run as v_wmma_f32_16x16x32_bf16
//   with fp32 accumulation. Weights are pre-swizzled to bf16 WMMA B-fragment
//   layout in d_ws by a prep kernel. drift uses the factored form
//   AU g - AU^T g + AL (AL^T g) + 0.1 g (no 32x32x32 per-sample GEMM).
// ---------------------------------------------------------------------------

typedef __attribute__((ext_vector_type(16))) __bf16       v16bf;
typedef __attribute__((ext_vector_type(8)))  float        v8f;
typedef __attribute__((ext_vector_type(4)))  unsigned int u32x4;

struct B256 { u32x4 lo, hi; };

#define POT_BETA  0.5f
#define ONS_MIN_D 0.1f
#define DT_C      0.01f
#define SQRT_DT   0.1f

static __device__ __forceinline__ unsigned short f2b(float f) {
    unsigned int u = __float_as_uint(f);
    unsigned int r = u + 0x7FFFu + ((u >> 16) & 1u);   // round-to-nearest-even
    return (unsigned short)(r >> 16);
}
static __device__ __forceinline__ float b2f(unsigned short h) {
    return __uint_as_float(((unsigned int)h) << 16);
}
static __device__ __forceinline__ v16bf make_frag(u32x4 lo, u32x4 hi) {
    B256 t{lo, hi};
    return __builtin_bit_cast(v16bf, t);
}
// A fragment (16x32 bf16, MxK): lane row = lane&15; halves [o,o+8) and
// [o+16,o+24) with o = (lane>=16 ? 8 : 0)  -> two ds_load_b128
static __device__ __forceinline__ v16bf load_a(const unsigned short* rowBase,
                                               int lane, int kOff) {
    const int o = (lane & 16) ? 8 : 0;
    const u32x4* p = (const u32x4*)(rowBase + kOff + o);
    return make_frag(p[0], p[2]);
}
// B fragment (32x16 bf16, KxN), pre-swizzled: lane holds 16 contiguous halves
static __device__ __forceinline__ v16bf load_b(const unsigned short* base,
                                               int frag, int lane) {
    const u32x4* p = (const u32x4*)(base + frag * 512 + lane * 16);
    return make_frag(p[0], p[1]);
}
static __device__ __forceinline__ v8f wmma_bf(v16bf a, v16bf b, v8f c) {
    return __builtin_amdgcn_wmma_f32_16x16x32_bf16(
        /*neg_a=*/false, a, /*neg_b=*/false, b,
        /*c_mod=*/(short)0, c, /*reuse_a=*/false, /*reuse_b=*/false);
}
static __device__ __forceinline__ float fact(float u) {
    float r  = fmaxf(u, 0.0f);
    float r2 = fmaxf(u - 0.5f, 0.0f);
    return r * r - r2 * r2;
}
static __device__ __forceinline__ float factp(float u) {
    float r  = fmaxf(u, 0.0f);
    float r2 = fmaxf(u - 0.5f, 0.0f);
    return 2.0f * (r - r2);
}

// ---------------------------------------------------------------------------
// Prep: pack a weight matrix into bf16 WMMA B-fragments.
//   trans==0: B[k][n] = W[k][n]   (W row-major [Ktot][Ntot])
//   trans==1: B[k][n] = W[n][k]   (W row-major [Ntot][Ktot])
// Fragment index = nt*nKt + kt; element (lane,j): K = kt*32 + (j&7) + 16*(j>=8) + 8*(lane>=16),
// N = nt*16 + (lane&15). Stored lane-major so main kernel loads 32B/lane.
// ---------------------------------------------------------------------------
__global__ void pack_b_frags(const float* __restrict__ W, int Ktot, int Ntot,
                             int trans, int nKt, int nNt,
                             unsigned short* __restrict__ dst) {
    int tid = blockIdx.x * blockDim.x + threadIdx.x;
    int total = nKt * nNt * 512;
    if (tid >= total) return;
    int j    = tid & 15;
    int lane = (tid >> 4) & 31;
    int f    = tid >> 9;           // f = nt*nKt + kt
    int kt   = f % nKt;
    int nt   = f / nKt;
    int kk   = (j & 7) + ((j & 8) ? 16 : 0) + ((lane & 16) ? 8 : 0);
    int nn   = lane & 15;
    int Kg   = kt * 32 + kk;
    int Ng   = nt * 16 + nn;
    float v  = trans ? W[Ng * Ktot + Kg] : W[Kg * Ntot + Ng];
    dst[f * 512 + lane * 16 + j] = f2b(v);
}

// ---------------------------------------------------------------------------
// Fused main kernel: one 16-sample M-tile per block (128 threads = 4 waves).
// ---------------------------------------------------------------------------
__global__ __launch_bounds__(128)
void onsager_fused(const float* __restrict__ z0, const float* __restrict__ noise,
                   const float* __restrict__ b0, const float* __restrict__ b1,
                   const float* __restrict__ bm, const float* __restrict__ bp,
                   const float* __restrict__ bpl, const float* __restrict__ sigma,
                   const unsigned short* __restrict__ wsW0T,
                   const unsigned short* __restrict__ wsW1T,
                   const unsigned short* __restrict__ wsWPT,
                   const unsigned short* __restrict__ wsWPLT,
                   const unsigned short* __restrict__ wsWP,
                   const unsigned short* __restrict__ wsWPLB,
                   const unsigned short* __restrict__ wsW1B,
                   const unsigned short* __restrict__ wsW0B,
                   const unsigned short* __restrict__ wsWM,
                   float* __restrict__ out) {
    __shared__ unsigned short zb[16 * 32];      // z tile (bf16, A operand)
    __shared__ unsigned short h1b[16 * 128];    // h1 = f(u1)
    __shared__ unsigned short fp1b[16 * 128];   // f'(u1)  -> later s = dh1*f'(u1)
    __shared__ unsigned short hb[16 * 128];     // h = f(u2)+h1
    __shared__ unsigned short fp2b[16 * 128];   // f'(u2)  -> later t = dh*f'(u2)
    __shared__ unsigned short qb[16 * 32];      // q = 2*pot
    __shared__ unsigned short dhb[16 * 128];    // dh
    __shared__ unsigned short matAb[16 * 1024]; // matA (bf16)
    __shared__ float gf[16 * 32];               // g
    __shared__ float yf[16 * 32];               // y = AL^T g

    const int tid  = threadIdx.x;
    const int wv   = tid >> 5;
    const int lane = tid & 31;
    const int col  = lane & 15;
    const int rb   = (lane >> 4) << 3;          // +8 rows for upper half-wave
    const int m0   = blockIdx.x * 16;           // first sample of tile

    // ---- stage 0: stage z0 tile as bf16 --------------------------------
    for (int e = tid; e < 512; e += 128)
        zb[e] = f2b(z0[m0 * 32 + e]);
    __syncthreads();

    // ---- stage 1: u1 = z@W0^T + b0 ; h1 = f(u1), f'(u1) ----------------
    for (int it = 0; it < 2; ++it) {
        const int nt = wv * 2 + it;             // 8 N-tiles over H=128
        v8f c = {};
        v16bf a = load_a(&zb[(lane & 15) * 32], lane, 0);
        c = wmma_bf(a, load_b(wsW0T, nt, lane), c);
        const float bias = b0[nt * 16 + col];
#pragma unroll
        for (int v = 0; v < 8; ++v) {
            float u = c[v] + bias;
            int idx = (v + rb) * 128 + nt * 16 + col;
            h1b[idx]  = f2b(fact(u));
            fp1b[idx] = f2b(factp(u));
        }
    }
    __syncthreads();

    // ---- stage 2: u2 = h1@W1^T + b1 ; h = f(u2)+h1, f'(u2) -------------
    for (int it = 0; it < 2; ++it) {
        const int nt = wv * 2 + it;
        v8f c = {};
#pragma unroll
        for (int kt = 0; kt < 4; ++kt) {
            v16bf a = load_a(&h1b[(lane & 15) * 128], lane, kt * 32);
            c = wmma_bf(a, load_b(wsW1T, nt * 4 + kt, lane), c);
        }
        const float bias = b1[nt * 16 + col];
#pragma unroll
        for (int v = 0; v < 8; ++v) {
            float u = c[v] + bias;
            int idx = (v + rb) * 128 + nt * 16 + col;
            hb[idx]   = f2b(fact(u) + b2f(h1b[idx]));
            fp2b[idx] = f2b(factp(u));
        }
    }
    __syncthreads();

    // ---- stage 3: pot = h@Wp^T + z@Wpl^T + bp + bpl ; q = 2*pot --------
    if (wv < 2) {
        const int nt = wv;                      // 2 N-tiles over N=32
        v8f c = {};
#pragma unroll
        for (int kt = 0; kt < 4; ++kt) {
            v16bf a = load_a(&hb[(lane & 15) * 128], lane, kt * 32);
            c = wmma_bf(a, load_b(wsWPT, nt * 4 + kt, lane), c);
        }
        {
            v16bf a = load_a(&zb[(lane & 15) * 32], lane, 0);
            c = wmma_bf(a, load_b(wsWPLT, nt, lane), c);
        }
        const float bias = bp[nt * 16 + col] + bpl[nt * 16 + col];
#pragma unroll
        for (int v = 0; v < 8; ++v)
            qb[(v + rb) * 32 + nt * 16 + col] = f2b(2.0f * (c[v] + bias));
    }
    __syncthreads();

    // ---- stage 4: dh = q@Wp ; t = dh * f'(u2) (in place over fp2b) -----
    for (int it = 0; it < 2; ++it) {
        const int nt = wv * 2 + it;
        v8f c = {};
        v16bf a = load_a(&qb[(lane & 15) * 32], lane, 0);
        c = wmma_bf(a, load_b(wsWP, nt, lane), c);
#pragma unroll
        for (int v = 0; v < 8; ++v) {
            float dh = c[v];
            int idx = (v + rb) * 128 + nt * 16 + col;
            dhb[idx]  = f2b(dh);
            fp2b[idx] = f2b(dh * b2f(fp2b[idx]));
        }
    }
    __syncthreads();

    // ---- stage 5: dh1 = t@W1 + dh ; s = dh1 * f'(u1) (in place fp1b) ---
    for (int it = 0; it < 2; ++it) {
        const int nt = wv * 2 + it;
        v8f c = {};
#pragma unroll
        for (int kt = 0; kt < 4; ++kt) {
            v16bf a = load_a(&fp2b[(lane & 15) * 128], lane, kt * 32);
            c = wmma_bf(a, load_b(wsW1B, nt * 4 + kt, lane), c);
        }
#pragma unroll
        for (int v = 0; v < 8; ++v) {
            int idx = (v + rb) * 128 + nt * 16 + col;
            float dh1 = c[v] + b2f(dhb[idx]);
            fp1b[idx] = f2b(dh1 * b2f(fp1b[idx]));
        }
    }
    __syncthreads();

    // ---- stage 6: g = -(s@W0 + q@Wpl + 2*beta*z) -----------------------
    if (wv < 2) {
        const int nt = wv;
        v8f c = {};
#pragma unroll
        for (int kt = 0; kt < 4; ++kt) {
            v16bf a = load_a(&fp1b[(lane & 15) * 128], lane, kt * 32);
            c = wmma_bf(a, load_b(wsW0B, nt * 4 + kt, lane), c);
        }
        {
            v16bf a = load_a(&qb[(lane & 15) * 32], lane, 0);
            c = wmma_bf(a, load_b(wsWPLB, nt, lane), c);
        }
#pragma unroll
        for (int v = 0; v < 8; ++v) {
            int row = v + rb, n = nt * 16 + col;
            float x = z0[(m0 + row) * 32 + n];
            gf[row * 32 + n] = -(c[v] + 2.0f * POT_BETA * x);
        }
    }
    __syncthreads();

    // ---- stage 7: matA = h@Wm^T + bm (dominant GEMM, 64 N-tiles) -------
    for (int it = 0; it < 16; ++it) {
        const int nt = wv * 16 + it;
        if (it + 1 < 16)                        // prefetch next fragment group
            __builtin_prefetch(wsWM + (nt + 1) * 4 * 512, 0, 0);
        v8f c = {};
#pragma unroll
        for (int kt = 0; kt < 4; ++kt) {
            v16bf a = load_a(&hb[(lane & 15) * 128], lane, kt * 32);
            c = wmma_bf(a, load_b(wsWM, nt * 4 + kt, lane), c);
        }
        const float bias = bm[nt * 16 + col];
#pragma unroll
        for (int v = 0; v < 8; ++v)
            matAb[(v + rb) * 1024 + nt * 16 + col] = f2b(c[v] + bias);
    }
    __syncthreads();

    // ---- stage 8: y_k = sum_{i>=k} A[i][k] g_i  (AL^T g) ---------------
    {
        const int s = tid >> 3, kq = tid & 7;
        const unsigned short* A = &matAb[s * 1024];
        const float* g = &gf[s * 32];
#pragma unroll
        for (int kk = 0; kk < 4; ++kk) {
            int k = kq * 4 + kk;
            float y = 0.0f;
            for (int i = k; i < 32; ++i)
                y += b2f(A[i * 32 + k]) * g[i];
            yf[s * 32 + k] = y;
        }
    }
    __syncthreads();

    // ---- stage 9: drift = AL y + AU g - AU^T g + 0.1 g ; z1 ------------
    {
        const int s = tid >> 3, iq = tid & 7;
        const unsigned short* A = &matAb[s * 1024];
        const float* g = &gf[s * 32];
        const float* y = &yf[s * 32];
#pragma unroll
        for (int ii = 0; ii < 4; ++ii) {
            int i = iq * 4 + ii;
            float d = ONS_MIN_D * g[i];
            for (int k = 0; k <= i; ++k)        // (AL y)_i
                d += b2f(A[i * 32 + k]) * y[k];
            for (int j = i + 1; j < 32; ++j)    // (AU g)_i
                d += b2f(A[i * 32 + j]) * g[j];
            for (int j = 0; j < i; ++j)         // -(AU^T g)_i
                d -= b2f(A[j * 32 + i]) * g[j];
            int gi = (m0 + s) * 32 + i;
            out[gi] = z0[gi] + d * DT_C + noise[gi] * sigma[i] * SQRT_DT;
        }
    }
}

// ---------------------------------------------------------------------------
// Workspace layout (in bf16 halves); one fragment = 512 halves = 1 KB.
// ---------------------------------------------------------------------------
#define OFF_W0T  0u        //  8 frags: B[k][n]=W0[n][k]   K=32  N=128
#define OFF_W1T  4096u     // 32 frags: B[k][n]=W1[n][k]   K=128 N=128
#define OFF_WPT  20480u    //  8 frags: B[k][n]=Wp[n][k]   K=128 N=32
#define OFF_WPLT 24576u    //  2 frags: B[k][n]=Wpl[n][k]  K=32  N=32
#define OFF_WP   25600u    //  8 frags: B[k][n]=Wp[k][n]   K=32  N=128
#define OFF_WPLB 29696u    //  2 frags: B[k][n]=Wpl[k][n]  K=32  N=32
#define OFF_W1B  30720u    // 32 frags: B[k][n]=W1[k][n]   K=128 N=128
#define OFF_W0B  47104u    //  8 frags: B[k][n]=W0[k][n]   K=128 N=32
#define OFF_WM   51200u    // 256 frags: B[k][n]=Wm[n][k]  K=128 N=1024

extern "C" void kernel_launch(void* const* d_in, const int* in_sizes, int n_in,
                              void* d_out, int out_size, void* d_ws, size_t ws_size,
                              hipStream_t stream) {
    const float* z0    = (const float*)d_in[0];
    const float* noise = (const float*)d_in[1];
    const float* W0    = (const float*)d_in[2];
    const float* b0    = (const float*)d_in[3];
    const float* W1    = (const float*)d_in[4];
    const float* b1    = (const float*)d_in[5];
    const float* Wm    = (const float*)d_in[6];
    const float* bm    = (const float*)d_in[7];
    const float* Wp    = (const float*)d_in[8];
    const float* bp    = (const float*)d_in[9];
    const float* Wpl   = (const float*)d_in[10];
    const float* bpl   = (const float*)d_in[11];
    const float* sigma = (const float*)d_in[12];
    float* out = (float*)d_out;
    unsigned short* ws = (unsigned short*)d_ws;

    auto pack = [&](const float* W, int Kt, int Nt, int tr, int nk, int nn,
                    unsigned int off) {
        int total = nk * nn * 512;
        pack_b_frags<<<(total + 255) / 256, 256, 0, stream>>>(
            W, Kt, Nt, tr, nk, nn, ws + off);
    };
    pack(W0,  32,  128, 1, 1,  8,  OFF_W0T);   // forward: z @ W0^T
    pack(W1,  128, 128, 1, 4,  8,  OFF_W1T);   // forward: h1 @ W1^T
    pack(Wp,  128, 32,  1, 4,  2,  OFF_WPT);   // forward: h @ Wp^T
    pack(Wpl, 32,  32,  1, 1,  2,  OFF_WPLT);  // forward: z @ Wpl^T
    pack(Wp,  32,  128, 0, 1,  8,  OFF_WP);    // backward: q @ Wp
    pack(Wpl, 32,  32,  0, 1,  2,  OFF_WPLB);  // backward: q @ Wpl
    pack(W1,  128, 128, 0, 4,  8,  OFF_W1B);   // backward: t @ W1
    pack(W0,  128, 32,  0, 4,  2,  OFF_W0B);   // backward: s @ W0
    pack(Wm,  128, 1024,1, 4, 64,  OFF_WM);    // matA: h @ Wm^T

    const int Bn = in_sizes[0] / 32;            // batch size
    onsager_fused<<<Bn / 16, 128, 0, stream>>>(
        z0, noise, b0, b1, bm, bp, bpl, sigma,
        ws + OFF_W0T, ws + OFF_W1T, ws + OFF_WPT, ws + OFF_WPLT,
        ws + OFF_WP, ws + OFF_WPLB, ws + OFF_W1B, ws + OFF_W0B, ws + OFF_WM,
        out);

    (void)n_in; (void)out_size; (void)ws_size;
}
